// DecoderRNN_58987080843307
// MI455X (gfx1250) — matrix operand order repaired and checked by reading, not compiled, not run on hardware
//
#include <hip/hip_runtime.h>
#include <stdint.h>

// ---------------- problem constants ----------------
#define B_      8
#define P_      196      // 14*14 pixels
#define D_      2048
#define A_      700
#define E_      700
#define H_      700
#define V_      30000
#define TSTEPS  30
#define KCAT    (E_ + D_ + H_)   // 3448 : [x_emb | awe | h]
#define KCHUNK  512              // K chunk staged in LDS (16 k-blocks of 32)

typedef __attribute__((ext_vector_type(16))) __bf16 v16bf;
typedef __attribute__((ext_vector_type(8)))  float  v8f;

// ---------------- helpers ----------------
__device__ __forceinline__ uint32_t bf16pair(float lo, float hi) {
    // round-to-nearest-even fp32 -> bf16, packed (hi<<16)|lo
    uint32_t a = __float_as_uint(lo), b = __float_as_uint(hi);
    uint32_t al = (a + 0x7FFFu + ((a >> 16) & 1u)) >> 16;
    uint32_t bh = (b + 0x7FFFu + ((b >> 16) & 1u)) >> 16;
    return (bh << 16) | (al & 0xFFFFu);
}

__device__ __forceinline__ v16bf frag_cast(uint4 x, uint4 y) {
    union { uint4 q[2]; v16bf v; } u;
    u.q[0] = x; u.q[1] = y;
    return u.v;
}

__device__ __forceinline__ float sigm(float x) { return 1.0f / (1.0f + __expf(-x)); }

// ---------------- weight packing (once) ----------------
// Packs row-major fp32 W (Ktot x N), optionally the vertical concat of two
// matrices (for [W_ih;W_hh]), into bf16 WMMA B-fragment order:
//   word index = (((bn*nKblk + kb)*32 + lane)*8 + j)
//   lane: n = bn*16 + (lane&15), half = lane>>4
//   B 32x16 bf16 layout (contiguous k order, not ISA table order): lanes 0-15 hold K 0..15,
//   lanes 16-31 hold K 16..31; VGPR j holds K pair (half*16 + 2j, +1).
__global__ void pack_weights(const float* __restrict__ W1, int K1,
                             const float* __restrict__ W2, int K2, int N,
                             uint32_t* __restrict__ out, int nKblk, int nNtile) {
    int g = blockIdx.x * blockDim.x + threadIdx.x;
    int total = nKblk * nNtile * 32;
    if (g >= total) return;
    int bn   = g / (nKblk * 32);
    int rem  = g - bn * (nKblk * 32);
    int kb   = rem >> 5;
    int lane = rem & 31;
    int n    = bn * 16 + (lane & 15);
    int half = lane >> 4;
    uint32_t* dst = out + (size_t)g * 8;
#pragma unroll
    for (int j = 0; j < 8; ++j) {
        int k0 = kb * 32 + half * 16 + j * 2;
        float f0 = 0.f, f1 = 0.f;
        if (n < N) {
            int k = k0;
            if (k < K1)           f0 = W1[(size_t)k * N + n];
            else if (k - K1 < K2) f0 = W2[(size_t)(k - K1) * N + n];
            k = k0 + 1;
            if (k < K1)           f1 = W1[(size_t)k * N + n];
            else if (k - K1 < K2) f1 = W2[(size_t)(k - K1) * N + n];
        }
        dst[j] = bf16pair(f0, f1);
    }
}

// ---------------- generic WMMA GEMM ----------------
// C[Mrows x N] = A[Mrows x K](fp32,row-major,lda) @ Wpacked + bias
// REQUIRES: K and lda even (true for all call sites: 2048 / 700 / 3448).
// Block = 256 threads = 8 waves; wave w -> n-tile blockIdx.x*8+w, m-tile blockIdx.y.
// A chunk (16 x KCHUNK) converted to bf16 pairs in LDS in A-fragment order via
// single predicated float2 loads. Inner loop is software-pipelined (register
// double-buffer): steady state loads k-block kb's frags (linear addresses)
// while the WMMA for kb-1 executes; one epilogue WMMA consumes the final frags.
__global__ void __launch_bounds__(256)
gemm_wmma_bf16(const float* __restrict__ Ain, int lda, int Mrows, int K,
               const uint32_t* __restrict__ pk, int nKblk,
               const float* __restrict__ bias,
               float* __restrict__ C, long long ldc, int N) {
    __shared__ __align__(16) uint32_t As[16 * (KCHUNK / 2)];   // 16 KB
    const int tid   = threadIdx.x;
    const int lane  = tid & 31;
    const int wave  = tid >> 5;
    const int mtile = blockIdx.y;
    const int ntile = blockIdx.x * 8 + wave;
    const bool active = (ntile * 16) < N;

    v8f acc = {0.f, 0.f, 0.f, 0.f, 0.f, 0.f, 0.f, 0.f};
    const int totalK = nKblk * 32;
    const int m    = lane & 15;
    const int half = lane >> 4;
    const int aq0  = m * (KCHUNK / 8) + half * 2;   // uint4 index of A frag, kb=0

    for (int k0 = 0; k0 < totalK; k0 += KCHUNK) {
        __syncthreads();
        // cooperative load + fp32->bf16 convert of the A chunk.
        // K is even and k is even, so (k < K) <=> (k+1 < K): one float2 load.
        for (int idx = tid; idx < 16 * (KCHUNK / 2); idx += 256) {
            int mr  = idx >> 8;               // / (KCHUNK/2)
            int kk2 = idx & 255;
            int row = mtile * 16 + mr;
            int k   = k0 + kk2 * 2;
            float2 f = make_float2(0.f, 0.f);
            if (row < Mrows && k < K)
                f = *(const float2*)(Ain + (size_t)row * lda + k);
            As[idx] = bf16pair(f.x, f.y);
        }
        __syncthreads();
        if (active) {
            int rem   = nKblk - (k0 >> 5);
            int kbMax = rem < (KCHUNK / 32) ? rem : (KCHUNK / 32);
            const uint4* As4 = (const uint4*)As;
            // single B pointer, bumped 1 KB (64 uint4) per k-block
            const uint4* Bq =
                (const uint4*)(pk + ((size_t)(ntile * nKblk + (k0 >> 5)) * 32 + lane) * 8);

            // prologue: frags for kb = 0
            uint4 a0 = As4[aq0];
            uint4 a1 = As4[aq0 + 1];
            uint4 b0 = Bq[0];
            uint4 b1 = Bq[1];

            // steady state: load kb's frags (linear addresses), WMMA kb-1
            for (int kb = 1; kb < kbMax; ++kb) {
                Bq += 64;
                uint4 na0 = As4[aq0 + kb * 4];
                uint4 na1 = As4[aq0 + kb * 4 + 1];
                uint4 nb0 = Bq[0];
                uint4 nb1 = Bq[1];
                __builtin_prefetch((const void*)(Bq + 64), 0, 3);  // 1 KB ahead

                v16bf af = frag_cast(a0, a1);
                v16bf bf = frag_cast(b0, b1);
                acc = __builtin_amdgcn_wmma_f32_16x16x32_bf16(
                    false, af, false, bf, (short)0, acc, false, false);

                a0 = na0; a1 = na1; b0 = nb0; b1 = nb1;
            }
            // epilogue: last k-block of this chunk
            {
                v16bf af = frag_cast(a0, a1);
                v16bf bf = frag_cast(b0, b1);
                acc = __builtin_amdgcn_wmma_f32_16x16x32_bf16(
                    false, af, false, bf, (short)0, acc, false, false);
            }
        }
    }

    if (active) {
        int nCol = ntile * 16 + (lane & 15);
        int mOff = (lane >> 4) * 8;           // lanes>=16 hold M=8..15
        if (nCol < N) {
            float bv = bias ? bias[nCol] : 0.f;
#pragma unroll
            for (int v = 0; v < 8; ++v) {
                int row = mtile * 16 + mOff + v;
                if (row < Mrows)
                    C[(size_t)row * ldc + nCol] = acc[v] + bv;
            }
        }
    }
}

// ---------------- small supporting kernels ----------------
__global__ void mean_enc_kernel(const float* __restrict__ enc, float* __restrict__ me) {
    int b = blockIdx.x, tid = threadIdx.x;
    for (int d = tid; d < D_; d += 256) {
        float s = 0.f;
        for (int p = 0; p < P_; ++p) s += enc[((size_t)b * P_ + p) * D_ + d];
        me[(size_t)b * D_ + d] = s * (1.0f / P_);
    }
}

__global__ void vec_add2(const float* a, const float* b, float* o, int n) {
    int i = blockIdx.x * blockDim.x + threadIdx.x;
    if (i < n) o[i] = a[i] + b[i];
}

__global__ void init_state(const float* __restrict__ h, float* __restrict__ Xcat,
                           int* __restrict__ prev) {
    int b = blockIdx.x, tid = threadIdx.x;
    if (tid == 0) prev[b] = 1;  // reference: preds initialized to ones
    for (int j = tid; j < H_; j += 256)
        Xcat[(size_t)b * KCAT + (E_ + D_) + j] = h[(size_t)b * H_ + j];
}

// e = relu(att1 + att2)@W_full + b_full ; alpha = softmax_P(e) ; awe = alpha @ enc
__global__ void attention_kernel(const float* __restrict__ att1, const float* __restrict__ att2,
                                 const float* __restrict__ Wfull, const float* __restrict__ bfull,
                                 const float* __restrict__ enc,
                                 float* __restrict__ alphas_out, float* __restrict__ awe, int t) {
    int b = blockIdx.x, tid = threadIdx.x;
    __shared__ float e_s[P_];
    __shared__ float red[256];
    float ex = 0.f;
    if (tid < P_) {
        const float* a1 = att1 + ((size_t)b * P_ + tid) * A_;
        const float* a2 = att2 + (size_t)b * A_;
        float s = 0.f;
        for (int a = 0; a < A_; ++a) {
            float x = a1[a] + a2[a];
            s += (x > 0.f ? x : 0.f) * Wfull[a];
        }
        e_s[tid] = s + bfull[0];
    }
    __syncthreads();
    float mx = -3.4e38f;
    for (int p = tid; p < P_; p += 256) mx = fmaxf(mx, e_s[p]);
    red[tid] = mx; __syncthreads();
    for (int s = 128; s > 0; s >>= 1) { if (tid < s) red[tid] = fmaxf(red[tid], red[tid + s]); __syncthreads(); }
    mx = red[0]; __syncthreads();
    if (tid < P_) ex = __expf(e_s[tid] - mx);
    red[tid] = (tid < P_) ? ex : 0.f; __syncthreads();
    for (int s = 128; s > 0; s >>= 1) { if (tid < s) red[tid] += red[tid + s]; __syncthreads(); }
    float inv = 1.0f / red[0]; __syncthreads();
    if (tid < P_) {
        float al = ex * inv;
        e_s[tid] = al;
        alphas_out[((size_t)b * TSTEPS + t) * P_ + tid] = al;
    }
    __syncthreads();
    for (int d = tid; d < D_; d += 256) {
        float s = 0.f;
        for (int p = 0; p < P_; ++p) s += e_s[p] * enc[((size_t)b * P_ + p) * D_ + d];
        awe[(size_t)b * D_ + d] = s;
    }
}

// x = [ emb[prev] | sigmoid(h@W_fbeta + b_fbeta) * awe ]  (first E_+D_ cols of Xcat)
__global__ void gate_concat(const float* __restrict__ emb, const int* __restrict__ prev,
                            const float* __restrict__ fbet, const float* __restrict__ awe,
                            float* __restrict__ Xcat) {
    int b = blockIdx.x, tid = threadIdx.x;
    int pv = prev[b];
    for (int j = tid; j < E_ + D_; j += 256) {
        float v;
        if (j < E_) v = emb[(size_t)pv * E_ + j];
        else {
            int d = j - E_;
            v = sigm(fbet[(size_t)b * D_ + d]) * awe[(size_t)b * D_ + d];
        }
        Xcat[(size_t)b * KCAT + j] = v;
    }
}

__global__ void lstm_kernel(const float* __restrict__ gates, float* __restrict__ h,
                            float* __restrict__ c, float* __restrict__ Xcat) {
    int b = blockIdx.x, tid = threadIdx.x;
    const float* g = gates + (size_t)b * 4 * H_;
    for (int j = tid; j < H_; j += 256) {
        float ig = sigm(g[j]);
        float fg = sigm(g[H_ + j]);
        float gg = tanhf(g[2 * H_ + j]);
        float og = sigm(g[3 * H_ + j]);
        float cn = fg * c[(size_t)b * H_ + j] + ig * gg;
        float hn = og * tanhf(cn);
        c[(size_t)b * H_ + j] = cn;
        h[(size_t)b * H_ + j] = hn;
        Xcat[(size_t)b * KCAT + (E_ + D_) + j] = hn;  // h feeds next step's gates GEMM
    }
}

// first-occurrence argmax over V (matches jnp.argmax)
__global__ void argmax_kernel(const float* __restrict__ pred, int t, int* __restrict__ prev) {
    int b = blockIdx.x, tid = threadIdx.x;
    const float* row = pred + ((size_t)b * TSTEPS + t) * V_;
    __shared__ float sv[256];
    __shared__ int   si[256];
    float bv = -3.4e38f; int bi = V_;
    for (int v = tid; v < V_; v += 256) {
        float x = row[v];
        if (x > bv) { bv = x; bi = v; }
    }
    sv[tid] = bv; si[tid] = bi; __syncthreads();
    for (int s = 128; s > 0; s >>= 1) {
        if (tid < s) {
            if (sv[tid + s] > sv[tid] || (sv[tid + s] == sv[tid] && si[tid + s] < si[tid])) {
                sv[tid] = sv[tid + s]; si[tid] = si[tid + s];
            }
        }
        __syncthreads();
    }
    if (tid == 0) prev[b] = si[0];
}

// ---------------- host orchestration ----------------
extern "C" void kernel_launch(void* const* d_in, const int* in_sizes, int n_in,
                              void* d_out, int out_size, void* d_ws, size_t ws_size,
                              hipStream_t stream) {
    (void)in_sizes; (void)n_in; (void)out_size; (void)ws_size;
    const float* encoder_out = (const float*)d_in[0];
    /* d_in[1] encoded_captions: only its length (T=31 -> 30 steps) matters */
    const float* emb       = (const float*)d_in[2];
    const float* W_enc_att = (const float*)d_in[3];
    const float* b_enc_att = (const float*)d_in[4];
    const float* W_dec_att = (const float*)d_in[5];
    const float* b_dec_att = (const float*)d_in[6];
    const float* W_full    = (const float*)d_in[7];
    const float* b_full    = (const float*)d_in[8];
    const float* W_ih      = (const float*)d_in[9];
    const float* b_ih      = (const float*)d_in[10];
    const float* W_hh      = (const float*)d_in[11];
    const float* b_hh      = (const float*)d_in[12];
    const float* W_init_h  = (const float*)d_in[13];
    const float* b_init_h  = (const float*)d_in[14];
    const float* W_init_c  = (const float*)d_in[15];
    const float* b_init_c  = (const float*)d_in[16];
    const float* W_fbeta   = (const float*)d_in[17];
    const float* b_fbeta   = (const float*)d_in[18];
    const float* W_fc      = (const float*)d_in[19];
    const float* b_fc      = (const float*)d_in[20];

    float* pred_out  = (float*)d_out;                                  // (B,30,V)
    float* alpha_out = pred_out + (size_t)B_ * TSTEPS * V_;            // (B,30,P)

    // ---- workspace layout (≈80 MB; packed bf16 weights stay L2-resident) ----
    char* ws = (char*)d_ws;
    size_t off = 0;
    auto alloc = [&](size_t bytes) -> char* {
        char* p = ws + off;
        off += (bytes + 255) & ~(size_t)255;
        return p;
    };
    auto nkb  = [](int K) { return (K + 31) / 32; };
    auto nnt  = [](int N) { return (N + 15) / 16; };
    auto pksz = [&](int K, int N) { return (size_t)nkb(K) * nnt(N) * 256 * sizeof(uint32_t); };

    uint32_t* pk_enc    = (uint32_t*)alloc(pksz(D_, A_));
    uint32_t* pk_init_h = (uint32_t*)alloc(pksz(D_, H_));
    uint32_t* pk_init_c = (uint32_t*)alloc(pksz(D_, H_));
    uint32_t* pk_dec    = (uint32_t*)alloc(pksz(H_, A_));
    uint32_t* pk_fbeta  = (uint32_t*)alloc(pksz(H_, D_));
    uint32_t* pk_cat    = (uint32_t*)alloc(pksz(KCAT, 4 * H_));
    uint32_t* pk_fc     = (uint32_t*)alloc(pksz(H_, V_));

    float* att1     = (float*)alloc((size_t)B_ * P_ * A_ * 4);
    float* mean_enc = (float*)alloc((size_t)B_ * D_ * 4);
    float* h_buf    = (float*)alloc((size_t)B_ * H_ * 4);
    float* c_buf    = (float*)alloc((size_t)B_ * H_ * 4);
    float* att2     = (float*)alloc((size_t)B_ * A_ * 4);
    float* fbet     = (float*)alloc((size_t)B_ * D_ * 4);
    float* awe      = (float*)alloc((size_t)B_ * D_ * 4);
    float* Xcat     = (float*)alloc((size_t)B_ * KCAT * 4);
    float* gates    = (float*)alloc((size_t)B_ * 4 * H_ * 4);
    float* bias_cat = (float*)alloc((size_t)4 * H_ * 4);
    int*   prev     = (int*)alloc(B_ * sizeof(int));

    auto pack = [&](const float* W1, int K1, const float* W2, int K2, int N, uint32_t* dst) {
        int nk = nkb(K1 + K2), nn = nnt(N);
        int total = nk * nn * 32;
        pack_weights<<<(total + 255) / 256, 256, 0, stream>>>(W1, K1, W2, K2, N, dst, nk, nn);
    };
    auto gemm = [&](const float* Ain, int lda, int Mrows, int K, const uint32_t* pkw,
                    const float* bias, float* C, long long ldc, int N) {
        int nk = nkb(K), nn = nnt(N);
        dim3 grid((nn + 7) / 8, (Mrows + 15) / 16);
        gemm_wmma_bf16<<<grid, 256, 0, stream>>>(Ain, lda, Mrows, K, pkw, nk, bias, C, ldc, N);
    };

    // ---- one-time prep: pack weights to bf16 WMMA fragment order ----
    pack(W_enc_att, D_, nullptr, 0, A_, pk_enc);
    pack(W_init_h,  D_, nullptr, 0, H_, pk_init_h);
    pack(W_init_c,  D_, nullptr, 0, H_, pk_init_c);
    pack(W_dec_att, H_, nullptr, 0, A_, pk_dec);
    pack(W_fbeta,   H_, nullptr, 0, D_, pk_fbeta);
    pack(W_ih, E_ + D_, W_hh, H_, 4 * H_, pk_cat);       // fused [W_ih;W_hh]
    pack(W_fc,      H_, nullptr, 0, V_, pk_fc);
    vec_add2<<<(4 * H_ + 255) / 256, 256, 0, stream>>>(b_ih, b_hh, bias_cat, 4 * H_);

    // ---- init: mean pooling, h0/c0, hoisted att1 ----
    mean_enc_kernel<<<B_, 256, 0, stream>>>(encoder_out, mean_enc);
    gemm(encoder_out, D_, B_ * P_, D_, pk_enc, b_enc_att, att1, A_, A_);   // att1 (B*P, A)
    gemm(mean_enc, D_, B_, D_, pk_init_h, b_init_h, h_buf, H_, H_);
    gemm(mean_enc, D_, B_, D_, pk_init_c, b_init_c, c_buf, H_, H_);
    init_state<<<B_, 256, 0, stream>>>(h_buf, Xcat, prev);

    // ---- 30 greedy decode steps ----
    for (int t = 0; t < TSTEPS; ++t) {
        gemm(h_buf, H_, B_, H_, pk_dec, b_dec_att, att2, A_, A_);
        attention_kernel<<<B_, 256, 0, stream>>>(att1, att2, W_full, b_full,
                                                 encoder_out, alpha_out, awe, t);
        gemm(h_buf, H_, B_, H_, pk_fbeta, b_fbeta, fbet, D_, D_);
        gate_concat<<<B_, 256, 0, stream>>>(emb, prev, fbet, awe, Xcat);
        gemm(Xcat, KCAT, B_, KCAT, pk_cat, bias_cat, gates, 4 * H_, 4 * H_);
        lstm_kernel<<<B_, 256, 0, stream>>>(gates, h_buf, c_buf, Xcat);
        gemm(h_buf, H_, B_, H_, pk_fc, b_fc,
             pred_out + (size_t)t * V_, (long long)TSTEPS * V_, V_);      // logits -> output
        argmax_kernel<<<B_, 256, 0, stream>>>(pred_out, t, prev);
    }
}
